// CrossAttention_83648783057644
// MI455X (gfx1250) — compile-verified, matrix operand order checked by reading, not run on hardware
//
#include <hip/hip_runtime.h>
#include <hip/hip_bf16.h>
#include <stdint.h>

// CDNA5 / gfx1250: wave32, WMMA 16x16x32 bf16 (f32 accum), async loads to LDS.

typedef __attribute__((ext_vector_type(16))) __bf16 v16bf;
typedef __attribute__((ext_vector_type(8)))  float  v8f;

#define BATCH 4
#define NQDIM 1024
#define NKDIM 1024
#define CDIM  1024
#define HEADS 16
#define DHEAD 64

// Convert 16 consecutive f32 to a bf16 fragment (two 32B vector loads + cvts).
__device__ __forceinline__ v16bf load16_cvt_f32(const float* __restrict__ p) {
  v8f lo = *(const v8f*)p;
  v8f hi = *(const v8f*)(p + 8);
  v16bf r;
#pragma unroll
  for (int i = 0; i < 8; ++i) { r[i] = (__bf16)lo[i]; r[i + 8] = (__bf16)hi[i]; }
  return r;
}

__device__ __forceinline__ v8f wmma_bf16(v16bf a, v16bf b, v8f c) {
  // (neg_a, A, neg_b, B, c_mod, C, reuse_a, reuse_b)
  return __builtin_amdgcn_wmma_f32_16x16x32_bf16(false, a, false, b, (short)0, c,
                                                 false, false);
}

// CDNA5 async DMA: copy 16B from global memory into LDS, tracked by ASYNCcnt.
__device__ __forceinline__ void async_load_b128(uint32_t lds_addr,
                                                const void* gaddr) {
  asm volatile("global_load_async_to_lds_b128 %0, %1, off"
               :: "v"(lds_addr), "v"(gaddr)
               : "memory");
}

// ---------------------------------------------------------------------------
// GEMM: out[m,n] = sum_k x[m,k] * W[n,k] + bias[n]   (M=4096, N=K=1024)
// One wave computes a 32x32 block (4 WMMA tiles) for 2x data reuse and 4
// independent WMMAs per k-step. Output bf16, head-split:
//   transpose_out==0: out[((b*H+h)*N + q)*64 + d]      (Q, K)
//   transpose_out==1: out[((b*H+h)*64 + d)*NK + q]     (V, transposed for P*V)
// ---------------------------------------------------------------------------
__global__ __launch_bounds__(256) void qkv_proj_kernel(
    const float* __restrict__ x, const float* __restrict__ W,
    const float* __restrict__ bias, __bf16* __restrict__ out, int transpose_out) {
  const int lane = threadIdx.x;
  const int lo = lane & 15, hi = lane >> 4;
  const int tile = blockIdx.x * 8 + threadIdx.y;  // 4096 tiles (128 x 32)
  const int mt = tile >> 5;                       // 128 row blocks of 32
  const int nt = tile & 31;                       // 32 col blocks of 32
  const int m0 = mt * 32, n0 = nt * 32;

  const float* arow0 = x + (size_t)(m0 + lo) * CDIM + hi * 16;
  const float* arow1 = arow0 + (size_t)16 * CDIM;
  const float* brow0 = W + (size_t)(n0 + lo) * CDIM + hi * 16;
  const float* brow1 = brow0 + (size_t)16 * CDIM;

  v8f c00 = {}, c01 = {}, c10 = {}, c11 = {};
#pragma unroll 2
  for (int k = 0; k < CDIM; k += 32) {
    v16bf a0 = load16_cvt_f32(arow0 + k);
    v16bf a1 = load16_cvt_f32(arow1 + k);
    v16bf b0 = load16_cvt_f32(brow0 + k);
    v16bf b1 = load16_cvt_f32(brow1 + k);
    c00 = wmma_bf16(a0, b0, c00);
    c01 = wmma_bf16(a0, b1, c01);
    c10 = wmma_bf16(a1, b0, c10);
    c11 = wmma_bf16(a1, b1, c11);
  }

  auto store_tile = [&](v8f c, int ms, int ns) {
    const int n = ns + lo;
    const int h = n >> 6, d = n & 63;
    const float bn = bias[n];
    const int bidx = ms >> 10;  // 32-row block never straddles a batch
    const int q0 = ms & 1023;
    if (!transpose_out) {
#pragma unroll
      for (int v = 0; v < 8; ++v) {
        int q = q0 + v + 8 * hi;
        out[((size_t)(bidx * HEADS + h) * NQDIM + q) * DHEAD + d] =
            (__bf16)(c[v] + bn);
      }
    } else {
#pragma unroll
      for (int v = 0; v < 8; ++v) {
        int q = q0 + v + 8 * hi;
        out[((size_t)(bidx * HEADS + h) * DHEAD + d) * NKDIM + q] =
            (__bf16)(c[v] + bn);
      }
    }
  };
  store_tile(c00, m0, n0);
  store_tile(c01, m0, n0 + 16);
  store_tile(c10, m0 + 16, n0);
  store_tile(c11, m0 + 16, n0 + 16);
}

// ---------------------------------------------------------------------------
// Flash attention. All 8 waves of a block share one (b,h): K/V chunks are
// staged into LDS once per block with double-buffered async-to-LDS DMA
// (ASYNCcnt), overlapping the next chunk's copy with this chunk's WMMAs.
// One wave owns one 16-row q-tile; online softmax over NK in chunks of 32.
// ---------------------------------------------------------------------------
__global__ __launch_bounds__(256) void flash_attn_kernel(
    const __bf16* __restrict__ Qp, const __bf16* __restrict__ Kp,
    const __bf16* __restrict__ Vp, const float* __restrict__ bias,
    __bf16* __restrict__ Y) {
  __shared__ __align__(16) __bf16 kbuf[2][32 * DHEAD];  // 2 x 4KB  [krow][d]
  __shared__ __align__(16) __bf16 vbuf[2][DHEAD * 32];  // 2 x 4KB  [d][krow]
  __shared__ __align__(32) __bf16 plds[8][16][32];      // 8KB, per-wave P tile

  const int lane = threadIdx.x, w = threadIdx.y;
  const int tid = w * 32 + lane;
  const int lo = lane & 15, hi = lane >> 4;
  const int bh = blockIdx.x >> 3;                  // 8 blocks per (b,h)
  const int qt = ((blockIdx.x & 7) << 3) + w;      // q-tile of this wave
  const int b = bh >> 4, h = bh & 15;
  const int q0 = qt * 16;

  const __bf16* kgbase = Kp + (size_t)bh * NKDIM * DHEAD;  // [nk][64]
  const __bf16* vgbase = Vp + (size_t)bh * DHEAD * NKDIM;  // [64][nk]
  const int vrow = tid >> 2, vcol = (tid & 3) * 8;

  auto stage = [&](int kc, int buf) {
    // K chunk: rows kc..kc+31 are contiguous (2048 bf16 = 4KB); 256x16B.
    async_load_b128((uint32_t)(size_t)&kbuf[buf][tid * 8],
                    kgbase + (size_t)kc * DHEAD + tid * 8);
    // V chunk: 64 rows (d) x 32 cols (krow), row stride NK; 4 threads/row.
    async_load_b128((uint32_t)(size_t)&vbuf[buf][vrow * 32 + vcol],
                    vgbase + (size_t)vrow * NKDIM + kc + vcol);
  };

  // Q A-fragments for d = 0..31 and d = 32..63
  const __bf16* qbase = Qp + ((size_t)bh * NQDIM + q0 + lo) * DHEAD + hi * 16;
  v16bf qa0 = *(const v16bf*)(qbase);
  v16bf qa1 = *(const v16bf*)(qbase + 32);

  const float scale = 0.125f;  // 1/sqrt(64)
  float mrow[8], lrow[8];
  v8f o0 = {}, o1 = {}, o2 = {}, o3 = {};
#pragma unroll
  for (int v = 0; v < 8; ++v) { mrow[v] = -1e30f; lrow[v] = 0.0f; }

  const float* bias_base = bias + ((size_t)h * NQDIM + q0) * NKDIM;

  stage(0, 0);
  for (int kc = 0; kc < NKDIM; kc += 32) {
    const int cur = (kc >> 5) & 1;
    // Own async copies done; barrier makes every thread's chunk visible and
    // guarantees the previous chunk's readers have all finished.
    asm volatile("s_wait_asynccnt 0x0" ::: "memory");
    __syncthreads();
    if (kc + 32 < NKDIM) stage(kc + 32, cur ^ 1);  // overlap DMA with compute

    // ---- S = (Q K^T) * scale + bias : two 16x16 tiles (cols kc..kc+31)
    const __bf16* kb_base = &kbuf[cur][lo * DHEAD + hi * 16];
    v16bf kb00 = *(const v16bf*)(kb_base);
    v16bf kb01 = *(const v16bf*)(kb_base + 32);
    v16bf kb10 = *(const v16bf*)(kb_base + 16 * DHEAD);
    v16bf kb11 = *(const v16bf*)(kb_base + 16 * DHEAD + 32);
    v8f s0 = {}, s1 = {};
    s0 = wmma_bf16(qa0, kb00, s0);
    s0 = wmma_bf16(qa1, kb01, s0);
    s1 = wmma_bf16(qa0, kb10, s1);
    s1 = wmma_bf16(qa1, kb11, s1);

#pragma unroll
    for (int v = 0; v < 8; ++v) {
      const float* bp = bias_base + (size_t)(v + 8 * hi) * NKDIM + kc + lo;
      s0[v] = s0[v] * scale + bp[0];
      s1[v] = s1[v] * scale + bp[16];
    }

    // ---- online softmax (row stats live across 16-lane halves)
    float alpha[8];
#pragma unroll
    for (int v = 0; v < 8; ++v) {
      float t = fmaxf(s0[v], s1[v]);
      t = fmaxf(t, __shfl_xor(t, 1));
      t = fmaxf(t, __shfl_xor(t, 2));
      t = fmaxf(t, __shfl_xor(t, 4));
      t = fmaxf(t, __shfl_xor(t, 8));
      float mn = fmaxf(mrow[v], t);
      alpha[v] = __expf(mrow[v] - mn);
      mrow[v] = mn;
      s0[v] = __expf(s0[v] - mn);
      s1[v] = __expf(s1[v] - mn);
      float rs = s0[v] + s1[v];
      rs += __shfl_xor(rs, 1);
      rs += __shfl_xor(rs, 2);
      rs += __shfl_xor(rs, 4);
      rs += __shfl_xor(rs, 8);
      lrow[v] = lrow[v] * alpha[v] + rs;
    }
#pragma unroll
    for (int v = 0; v < 8; ++v) {
      o0[v] *= alpha[v]; o1[v] *= alpha[v];
      o2[v] *= alpha[v]; o3[v] *= alpha[v];
    }

    // ---- P (C-layout) -> LDS -> A-fragment (wave-local; DS ops in-order,
    //      s_wait_dscnt also stops the compiler hoisting the ds_loads)
#pragma unroll
    for (int v = 0; v < 8; ++v) {
      int r = v + 8 * hi;
      plds[w][r][lo] = (__bf16)s0[v];
      plds[w][r][lo + 16] = (__bf16)s1[v];
    }
    asm volatile("s_wait_dscnt 0x0" ::: "memory");
    v16bf pa = *(const v16bf*)(&plds[w][lo][hi * 16]);
    asm volatile("s_wait_dscnt 0x0" ::: "memory");

    // ---- O += P * V  (V staged transposed: [d][krow], contiguous B-frags)
    const __bf16* vb_base = &vbuf[cur][lo * 32 + hi * 16];
    v16bf vb0 = *(const v16bf*)(vb_base);
    v16bf vb1 = *(const v16bf*)(vb_base + 16 * 32);
    v16bf vb2 = *(const v16bf*)(vb_base + 32 * 32);
    v16bf vb3 = *(const v16bf*)(vb_base + 48 * 32);
    o0 = wmma_bf16(pa, vb0, o0);
    o1 = wmma_bf16(pa, vb1, o1);
    o2 = wmma_bf16(pa, vb2, o2);
    o3 = wmma_bf16(pa, vb3, o3);
  }

  // ---- normalize, merge heads: Y[b, q, h*64 + col]
#pragma unroll
  for (int v = 0; v < 8; ++v) {
    float inv = 1.0f / lrow[v];
    int q = q0 + v + 8 * hi;
    __bf16* yrow = Y + ((size_t)b * NQDIM + q) * CDIM + h * DHEAD;
    yrow[lo] = (__bf16)(o0[v] * inv);
    yrow[16 + lo] = (__bf16)(o1[v] * inv);
    yrow[32 + lo] = (__bf16)(o2[v] * inv);
    yrow[48 + lo] = (__bf16)(o3[v] * inv);
  }
}

// ---------------------------------------------------------------------------
// Output projection: out[m,n] = sum_k Y[m,k] * Wp[n,k] + bp[n]  (f32 output)
// 32x32 block per wave, same blocking as qkv_proj.
// ---------------------------------------------------------------------------
__global__ __launch_bounds__(256) void out_proj_kernel(
    const __bf16* __restrict__ Y, const float* __restrict__ W,
    const float* __restrict__ bias, float* __restrict__ out) {
  const int lane = threadIdx.x;
  const int lo = lane & 15, hi = lane >> 4;
  const int tile = blockIdx.x * 8 + threadIdx.y;  // 4096 tiles (128 x 32)
  const int mt = tile >> 5, nt = tile & 31;
  const int m0 = mt * 32, n0 = nt * 32;

  const __bf16* arow0 = Y + (size_t)(m0 + lo) * CDIM + hi * 16;
  const __bf16* arow1 = arow0 + (size_t)16 * CDIM;
  const float* brow0 = W + (size_t)(n0 + lo) * CDIM + hi * 16;
  const float* brow1 = brow0 + (size_t)16 * CDIM;

  v8f c00 = {}, c01 = {}, c10 = {}, c11 = {};
#pragma unroll 2
  for (int k = 0; k < CDIM; k += 32) {
    v16bf a0 = *(const v16bf*)(arow0 + k);
    v16bf a1 = *(const v16bf*)(arow1 + k);
    v16bf b0 = load16_cvt_f32(brow0 + k);
    v16bf b1 = load16_cvt_f32(brow1 + k);
    c00 = wmma_bf16(a0, b0, c00);
    c01 = wmma_bf16(a0, b1, c01);
    c10 = wmma_bf16(a1, b0, c10);
    c11 = wmma_bf16(a1, b1, c11);
  }

  auto store_tile = [&](v8f c, int ms, int ns) {
    const int n = ns + lo;
    const float bn = bias[n];
#pragma unroll
    for (int v = 0; v < 8; ++v) {
      out[(size_t)(ms + v + 8 * hi) * CDIM + n] = c[v] + bn;
    }
  };
  store_tile(c00, m0, n0);
  store_tile(c01, m0, n0 + 16);
  store_tile(c10, m0 + 16, n0);
  store_tile(c11, m0 + 16, n0 + 16);
}

extern "C" void kernel_launch(void* const* d_in, const int* in_sizes, int n_in,
                              void* d_out, int out_size, void* d_ws, size_t ws_size,
                              hipStream_t stream) {
  (void)in_sizes; (void)n_in; (void)out_size; (void)ws_size;
  const float* query     = (const float*)d_in[0];
  const float* key       = (const float*)d_in[1];
  const float* attn_bias = (const float*)d_in[2];
  const float* Wq = (const float*)d_in[3];
  const float* bq = (const float*)d_in[4];
  const float* Wk = (const float*)d_in[5];
  const float* bk = (const float*)d_in[6];
  const float* Wv = (const float*)d_in[7];
  const float* bv = (const float*)d_in[8];
  const float* Wp = (const float*)d_in[9];
  const float* bp = (const float*)d_in[10];
  float* out = (float*)d_out;

  const size_t elems = (size_t)BATCH * HEADS * NQDIM * DHEAD;  // 4 Mi
  __bf16* Qp = (__bf16*)d_ws;
  __bf16* Kp = Qp + elems;
  __bf16* Vp = Kp + elems;
  __bf16* Yb = Vp + elems;

  dim3 blk(32, 8);  // 8 wave32 waves
  qkv_proj_kernel<<<512, blk, 0, stream>>>(query, Wq, bq, Qp, 0);
  qkv_proj_kernel<<<512, blk, 0, stream>>>(key,   Wk, bk, Kp, 0);
  qkv_proj_kernel<<<512, blk, 0, stream>>>(key,   Wv, bv, Vp, 1);
  flash_attn_kernel<<<512, blk, 0, stream>>>(Qp, Kp, Vp, attn_bias, Yb);
  out_proj_kernel<<<512, blk, 0, stream>>>(Yb, Wp, bp, out);
}
